// TrainerRNN_20598663151763
// MI455X (gfx1250) — matrix operand order repaired and checked by reading, not compiled, not run on hardware
//
#include <hip/hip_runtime.h>
#include <hip/hip_bf16.h>
#include <math.h>

// ---------------------------------------------------------------------------
// GRU stack (10 layers, T=2048, IN=384, H=768) for MI455X / gfx1250.
//
//  * GEMM-shaped work (per-layer gi = X W_ih^T + b_ih, final FC head) runs on
//    v_wmma_f32_16x16x32_bf16, fp32 accumulate, bf16 operands.
//  * The sequential recurrence runs in a 16-workgroup kernel per layer:
//    WG g owns h-indices [48g, 48g+48) and keeps its 144 W_hh rows (3 gates)
//    LDS-resident in bf16 (~221 KB), staged by the Tensor Data Mover
//    (tensor_load_to_lds + s_wait_tensorcnt).  Only the 768-float h vector
//    crosses WGs per step (L2 + monotonic-counter grid barrier, which needs an
//    initial counter value that is a multiple of 16).
// ---------------------------------------------------------------------------

#define T_SEQ 2048
#define IN_SZ 384
#define H_SZ  768
#define G3    2304      // 3*H
#define NLAY  10
#define NOUT  384

typedef __attribute__((ext_vector_type(16))) __bf16 v16bf;
typedef __attribute__((ext_vector_type(8)))  __bf16 v8bf;
typedef __attribute__((ext_vector_type(8)))  float  v8f;
typedef __attribute__((ext_vector_type(4)))  unsigned tdm_g0_t;
typedef __attribute__((ext_vector_type(8)))  unsigned tdm_g1_t;

// ---- bf16 helpers (bit-level) ---------------------------------------------
__device__ __forceinline__ float bflo(unsigned u) {           // low bf16 -> f32
    return __builtin_bit_cast(float, u << 16);
}
__device__ __forceinline__ float bfhi(unsigned u) {           // high bf16 -> f32
    return __builtin_bit_cast(float, u & 0xffff0000u);
}
__device__ __forceinline__ unsigned short f2bf(float f) {     // RNE
    unsigned u = __builtin_bit_cast(unsigned, f);
    u += 0x7fffu + ((u >> 16) & 1u);
    return (unsigned short)(u >> 16);
}

// ---- fp32 -> bf16 pack -----------------------------------------------------
__global__ void pack_bf16(const float* __restrict__ s,
                          unsigned short* __restrict__ d, int n) {
    int i = blockIdx.x * 256 + threadIdx.x;
    if (i < n) d[i] = f2bf(s[i]);
}

// ---------------------------------------------------------------------------
// WMMA GEMM:  C[M x N] = A[M x K](bf16) * W[N x K]^T(bf16) + bias[N]
// Block: 256 threads = 8 waves; block tile 64(M) x 128(N);
// wave tile 16(M) x 64(N) = 4 accumulators; K stepped by 32.
// A/B fragments use contiguous k = 16*half + s; C/D layout per CDNA5 ISA 7.12.2 (see round-1 notes).
// ---------------------------------------------------------------------------
__global__ __launch_bounds__(256) void wmma_gemm_bias(
    const unsigned short* __restrict__ A,   // [M x K] bf16 bits, row-major
    const unsigned short* __restrict__ W,   // [N x K] bf16 bits, row-major
    const float* __restrict__ bias,         // [N]
    float* __restrict__ C,                  // [M x N] fp32
    int M, int N, int K)
{
    const int lane = threadIdx.x & 31;
    const int wave = threadIdx.x >> 5;
    const int wm   = wave >> 1;              // 0..3
    const int wn   = wave & 1;               // 0..1
    const int row0 = blockIdx.y * 64 + wm * 16;
    const int col0 = blockIdx.x * 128 + wn * 64;
    const int half = lane >> 4;
    const int l16  = lane & 15;

    const unsigned short* aptr = A + (size_t)(row0 + l16) * K + 16 * half;

    v8f acc0 = {}, acc1 = {}, acc2 = {}, acc3 = {};

    for (int kk = 0; kk < K; kk += 32) {
        __builtin_prefetch(aptr + kk + 64, 0, 0);   // -> global_prefetch_b8
        v8bf a0 = *(const v8bf*)(aptr + kk);
        v8bf a1 = *(const v8bf*)(aptr + kk + 8);
        v16bf afrag = __builtin_shufflevector(
            a0, a1, 0,1,2,3,4,5,6,7,8,9,10,11,12,13,14,15);

        const int kb = kk + 16 * half;
        const unsigned short* w0 = W + (size_t)(col0 +  0 + l16) * K + kb;
        const unsigned short* w1 = W + (size_t)(col0 + 16 + l16) * K + kb;
        const unsigned short* w2 = W + (size_t)(col0 + 32 + l16) * K + kb;
        const unsigned short* w3 = W + (size_t)(col0 + 48 + l16) * K + kb;
        v16bf b0 = *(const v16bf*)w0;
        v16bf b1 = *(const v16bf*)w1;
        v16bf b2 = *(const v16bf*)w2;
        v16bf b3 = *(const v16bf*)w3;

        acc0 = __builtin_amdgcn_wmma_f32_16x16x32_bf16(false, afrag, false, b0,
                                                       (short)0, acc0, false, false);
        acc1 = __builtin_amdgcn_wmma_f32_16x16x32_bf16(false, afrag, false, b1,
                                                       (short)0, acc1, false, false);
        acc2 = __builtin_amdgcn_wmma_f32_16x16x32_bf16(false, afrag, false, b2,
                                                       (short)0, acc2, false, false);
        acc3 = __builtin_amdgcn_wmma_f32_16x16x32_bf16(false, afrag, false, b3,
                                                       (short)0, acc3, false, false);
    }

    v8f* accs[4] = { &acc0, &acc1, &acc2, &acc3 };
#pragma unroll
    for (int s = 0; s < 4; ++s) {
        const int col = col0 + 16 * s + l16;
        const float b = bias[col];
        const v8f a = *accs[s];
#pragma unroll
        for (int v = 0; v < 8; ++v) {
            const int r = row0 + v + 8 * half;
            C[(size_t)r * N + col] = a[v] + b;
        }
    }
}

// ---------------------------------------------------------------------------
// TDM: load a 2D tile (rows x row_elems of 2-byte elements, row stride
// row_stride_elems) from global memory into LDS at lds_byte_off.
// Descriptor layout per CDNA5 ISA ch.8 (D# groups 0/1, 2D tensor, no
// gather/iterate/pad/multicast, data_size=2B, type=2).
// ---------------------------------------------------------------------------
__device__ __forceinline__ void tdm_load_2d_to_lds(
    unsigned lds_byte_off, const void* gaddr,
    unsigned rows, unsigned row_elems, unsigned row_stride_elems)
{
    unsigned long long ga = (unsigned long long)(uintptr_t)gaddr;
    tdm_g0_t g0;
    g0[0] = 1u;                                        // count=1, user mode
    g0[1] = lds_byte_off;                              // lds_addr
    g0[2] = (unsigned)(ga & 0xffffffffu);              // global_addr[31:0]
    g0[3] = (unsigned)((ga >> 32) & 0x01ffffffu)       // global_addr[56:32]
          | (2u << 30);                                // type=2 ("image")
    tdm_g1_t g1;
    g1[0] = (1u << 16);                                // data_size=1 (2 bytes)
    g1[1] = (row_elems & 0xffffu) << 16;               // tensor_dim0[15:0]
    g1[2] = ((row_elems >> 16) & 0xffffu)              // tensor_dim0[31:16]
          | ((rows & 0xffffu) << 16);                  // tensor_dim1[15:0]
    g1[3] = ((rows >> 16) & 0xffffu)                   // tensor_dim1[31:16]
          | ((row_elems & 0xffffu) << 16);             // tile_dim0
    g1[4] = (rows & 0xffffu);                          // tile_dim1 (tile_dim2=0)
    g1[5] = row_stride_elems;                          // tensor_dim0_stride[31:0]
    g1[6] = 0u;                                        // stride hi / dim1_stride
    g1[7] = 0u;
    asm volatile("tensor_load_to_lds %0, %1" :: "s"(g0), "s"(g1) : "memory");
}

// ---------------------------------------------------------------------------
// Recurrent scan: 16 WGs x 256 threads.  WG g owns j in [48g, 48g+48).
// ---------------------------------------------------------------------------
#define SCAN_WGS 16
#define JS 48
#define DOTS 144            // 3 * JS

__device__ __forceinline__ void grid_sync(unsigned* bar) {
    __syncthreads();
    if (threadIdx.x == 0) {
        __threadfence();
        unsigned old = atomicAdd(bar, 1u);
        unsigned target = (old / SCAN_WGS + 1u) * SCAN_WGS;
        while (atomicAdd(bar, 0u) < target) {
            __builtin_amdgcn_s_sleep(1);
        }
    }
    __syncthreads();
}

__global__ __launch_bounds__(256) void gru_scan(
    const unsigned short* __restrict__ Whh,  // [2304 x 768] bf16 bits
    const float* __restrict__ bhh,           // [2304]
    const float* __restrict__ gi,            // [T x 2304] fp32
    float* h,                                // [768] global state (fp32)
    unsigned short* __restrict__ ybf,        // [T x 768] bf16 out
    unsigned* bar)
{
    __shared__ alignas(16) unsigned short s_w[DOTS * H_SZ];   // 221184 B
    __shared__ alignas(16) float s_h[H_SZ];                   //   3072 B
    __shared__ float s_gh[DOTS];

    const int g   = blockIdx.x;
    const int tid = threadIdx.x;
    const int j0  = g * JS;
    volatile float* hv = h;

    // Stage the 144 owned W_hh rows (3 blocks of 48 contiguous rows) into LDS
    // via the Tensor Data Mover.  Wave 0 issues; all waves sync on the result.
    if (tid < 32) {
        const unsigned lds_base = (unsigned)(uintptr_t)(void*)s_w;  // LDS byte offset
#pragma unroll
        for (int gate = 0; gate < 3; ++gate) {
            tdm_load_2d_to_lds(
                lds_base + (unsigned)(gate * JS * H_SZ * 2),
                Whh + (size_t)(gate * H_SZ + j0) * H_SZ,
                JS, H_SZ, H_SZ);
        }
        __builtin_amdgcn_s_wait_tensorcnt(0);
    }

    // h0 = 0 (each WG zeroes its slice), then sync so everyone sees it
    // (grid_sync also covers the LDS staging for all waves in this WG).
    if (tid < JS) hv[j0 + tid] = 0.0f;
    grid_sync(bar);

    for (int t = 0; t < T_SEQ; ++t) {
        // Pull full h into LDS.
        for (int i = tid; i < H_SZ; i += 256) s_h[i] = hv[i];
        __syncthreads();

        // 144 dot products of length 768, bf16 weights from LDS, fp32 acc.
        if (tid < DOTS) {
            const unsigned short* wrow = &s_w[tid * H_SZ];
            float acc0 = 0.0f, acc1 = 0.0f;
#pragma unroll 4
            for (int k = 0; k < H_SZ; k += 8) {
                uint4  wp = *(const uint4*)(wrow + k);
                float4 ha = *(const float4*)(s_h + k);
                float4 hb = *(const float4*)(s_h + k + 4);
                acc0 += bflo(wp.x) * ha.x;
                acc1 += bfhi(wp.x) * ha.y;
                acc0 += bflo(wp.y) * ha.z;
                acc1 += bfhi(wp.y) * ha.w;
                acc0 += bflo(wp.z) * hb.x;
                acc1 += bfhi(wp.z) * hb.y;
                acc0 += bflo(wp.w) * hb.z;
                acc1 += bfhi(wp.w) * hb.w;
            }
            const int gate = tid / JS;
            const int jj   = tid - gate * JS;
            s_gh[tid] = acc0 + acc1 + bhh[gate * H_SZ + j0 + jj];
        }
        __syncthreads();

        // Gate math + state update for this WG's 48 h-indices.
        if (tid < JS) {
            const int j = j0 + tid;
            const float ghr = s_gh[tid];
            const float ghz = s_gh[JS + tid];
            const float ghn = s_gh[2 * JS + tid];
            const float* git = gi + (size_t)t * G3;
            const float r = 1.0f / (1.0f + expf(-(git[j] + ghr)));
            const float z = 1.0f / (1.0f + expf(-(git[H_SZ + j] + ghz)));
            const float n = tanhf(git[2 * H_SZ + j] + r * ghn);
            const float hnew = (1.0f - z) * n + z * s_h[j];
            hv[j] = hnew;
            ybf[(size_t)t * H_SZ + j] = f2bf(hnew);
        }
        grid_sync(bar);
    }
}

// ---------------------------------------------------------------------------
// Host-side orchestration (all launches on `stream`, graph-capture safe).
// ---------------------------------------------------------------------------
extern "C" void kernel_launch(void* const* d_in, const int* in_sizes, int n_in,
                              void* d_out, int out_size, void* d_ws, size_t ws_size,
                              hipStream_t stream) {
    const float* x     = (const float*)d_in[0];
    const float* w_ih0 = (const float*)d_in[1];
    const float* w_hh0 = (const float*)d_in[2];
    const float* b_ih0 = (const float*)d_in[3];
    const float* b_hh0 = (const float*)d_in[4];
    const float* w_ih  = (const float*)d_in[5];   // [9, 2304, 768]
    const float* w_hh  = (const float*)d_in[6];   // [9, 2304, 768]
    const float* b_ih  = (const float*)d_in[7];   // [9, 2304]
    const float* b_hh  = (const float*)d_in[8];   // [9, 2304]
    const float* fc_w  = (const float*)d_in[9];   // [384, 768]
    const float* fc_b  = (const float*)d_in[10];  // [384]
    float* out = (float*)d_out;

    // Workspace carve-up (~94 MB total).
    char* ws = (char*)d_ws;
    size_t off = 0;
    auto take = [&](size_t bytes) -> char* {
        char* p = ws + off;
        off = (off + bytes + 255) & ~(size_t)255;
        return p;
    };
    unsigned short* xbf    = (unsigned short*)take((size_t)T_SEQ * IN_SZ * 2);
    unsigned short* wih0b  = (unsigned short*)take((size_t)G3 * IN_SZ * 2);
    unsigned short* whh0b  = (unsigned short*)take((size_t)G3 * H_SZ * 2);
    unsigned short* wihb   = (unsigned short*)take((size_t)9 * G3 * H_SZ * 2);
    unsigned short* whhb   = (unsigned short*)take((size_t)9 * G3 * H_SZ * 2);
    unsigned short* fcwb   = (unsigned short*)take((size_t)NOUT * H_SZ * 2);
    float*          gibuf  = (float*)take((size_t)T_SEQ * G3 * 4);
    unsigned short* ybf    = (unsigned short*)take((size_t)T_SEQ * H_SZ * 2);
    float*          hbuf   = (float*)take((size_t)H_SZ * 4);
    unsigned*       bar    = (unsigned*)take(256);
    (void)ws_size; (void)n_in; (void)in_sizes; (void)out_size;

    auto pack = [&](const float* s, unsigned short* d, int n) {
        pack_bf16<<<dim3((n + 255) / 256), dim3(256), 0, stream>>>(s, d, n);
    };
    pack(x,     xbf,   T_SEQ * IN_SZ);
    pack(w_ih0, wih0b, G3 * IN_SZ);
    pack(w_hh0, whh0b, G3 * H_SZ);
    pack(w_ih,  wihb,  9 * G3 * H_SZ);
    pack(w_hh,  whhb,  9 * G3 * H_SZ);
    pack(fc_w,  fcwb,  NOUT * H_SZ);

    // Layer 0: gi GEMM (K=384), then scan.
    wmma_gemm_bias<<<dim3(G3 / 128, T_SEQ / 64), dim3(256), 0, stream>>>(
        xbf, wih0b, b_ih0, gibuf, T_SEQ, G3, IN_SZ);
    gru_scan<<<dim3(SCAN_WGS), dim3(256), 0, stream>>>(
        whh0b, b_hh0, gibuf, hbuf, ybf, bar);

    // Layers 1..9: gi GEMM (K=768) from previous layer's bf16 output, then scan.
    for (int l = 1; l < NLAY; ++l) {
        const unsigned short* wihl = wihb + (size_t)(l - 1) * G3 * H_SZ;
        const unsigned short* whhl = whhb + (size_t)(l - 1) * G3 * H_SZ;
        const float* bihl = b_ih + (size_t)(l - 1) * G3;
        const float* bhhl = b_hh + (size_t)(l - 1) * G3;
        wmma_gemm_bias<<<dim3(G3 / 128, T_SEQ / 64), dim3(256), 0, stream>>>(
            ybf, wihl, bihl, gibuf, T_SEQ, G3, H_SZ);
        gru_scan<<<dim3(SCAN_WGS), dim3(256), 0, stream>>>(
            whhl, bhhl, gibuf, hbuf, ybf, bar);
    }

    // FC head: out = Y9 * fc_w^T + fc_b   (M=2048, N=384, K=768)
    wmma_gemm_bias<<<dim3(NOUT / 128, T_SEQ / 64), dim3(256), 0, stream>>>(
        ybf, fcwb, fc_b, out, T_SEQ, NOUT, H_SZ);
}
